// QuantizedLinear_24721831756585
// MI455X (gfx1250) — matrix operand order repaired and checked by reading, not compiled, not run on hardware
//
#include <hip/hip_runtime.h>

// ---------------- problem constants (from reference) ----------------
#define OUT_F   4096
#define IN_F    4096
#define GROUP   128
#define M_TOTAL 8192          // 4 * 2048 rows, batch flattened

// ---------------- tiling ----------------
#define BM 128                // M rows per workgroup
#define BN 64                 // N cols per workgroup
#define BK 32                 // K depth per stage (== one WMMA K)
#define NKT (IN_F / BK)       // 128 k-steps
#define LDA 40                // padded LDS row stride (ushorts); 80B = mult of 16
#define LDB 40

#define AS1 __attribute__((address_space(1)))
#define AS3 __attribute__((address_space(3)))

#if __has_builtin(__builtin_amdgcn_global_load_async_to_lds_b128) && \
    __has_builtin(__builtin_amdgcn_s_wait_asynccnt)
#define HAVE_ASYNC_LDS 1
#else
#define HAVE_ASYNC_LDS 0
#endif

typedef __attribute__((ext_vector_type(16))) __bf16 v16bf;
typedef __attribute__((ext_vector_type(2)))  __bf16 v2bf;
typedef __attribute__((ext_vector_type(8)))  float  v8f;
typedef __attribute__((ext_vector_type(4)))  int    v4i;   // b128 payload type

union Frag { v16bf v; uint4 q[2]; };   // 32 bytes

// fp32 -> bf16 (RNE) fallback
__device__ __forceinline__ unsigned int f2bf(float f) {
    union { float f; unsigned int u; } c; c.f = f;
    unsigned int u = c.u;
    u += 0x7FFFu + ((u >> 16) & 1u);
    return u >> 16;
}

// pack two fp32 -> one dword of 2x bf16; prefer the HW packed convert
__device__ __forceinline__ unsigned int pack_bf16(float lo, float hi) {
#if __has_builtin(__builtin_amdgcn_cvt_pk_bf16_f32)
    union { v2bf v; unsigned int u; } c;
    c.v = __builtin_amdgcn_cvt_pk_bf16_f32(lo, hi);
    return c.u;
#else
    return f2bf(lo) | (f2bf(hi) << 16);
#endif
}

// =====================================================================
// Phase 1a: x fp32 -> bf16 (done ONCE instead of once per N-tile column)
// =====================================================================
__global__ __launch_bounds__(256)
void convert_x_kernel(const float* __restrict__ x, unsigned short* __restrict__ xbf) {
    const size_t i = ((size_t)blockIdx.x * 256 + threadIdx.x) * 8;
    float4 f0 = *(const float4*)(x + i);
    float4 f1 = *(const float4*)(x + i + 4);
    uint4 o;
    o.x = pack_bf16(f0.x, f0.y);
    o.y = pack_bf16(f0.z, f0.w);
    o.z = pack_bf16(f1.x, f1.y);
    o.w = pack_bf16(f1.z, f1.w);
    *(uint4*)(xbf + i) = o;
}

// =====================================================================
// Phase 1b: dequant 4-bit W -> bf16 (done ONCE instead of once per M-tile row)
// =====================================================================
__global__ __launch_bounds__(256)
void dequant_w_kernel(const int*   __restrict__ packed,
                      const float* __restrict__ scales,
                      const float* __restrict__ offsets,
                      unsigned short* __restrict__ wbf) {
    const size_t tid = (size_t)blockIdx.x * 256 + threadIdx.x;
    const size_t e   = tid * 4;            // int32 element index (16B aligned)
    int4 q = *(const int4*)(packed + e);
    const size_t j = e * 2;                // flat code index (8 codes, one group)
    const int    g = (int)(j / GROUP);
    const float  s = scales[g];
    const float  o = offsets[g];
    uint4 r;
    r.x = pack_bf16((float)( q.x        & 15) * s + o,
                    (float)((q.x >> 4)  & 15) * s + o);
    r.y = pack_bf16((float)( q.y        & 15) * s + o,
                    (float)((q.y >> 4)  & 15) * s + o);
    r.z = pack_bf16((float)( q.z        & 15) * s + o,
                    (float)((q.z >> 4)  & 15) * s + o);
    r.w = pack_bf16((float)( q.w        & 15) * s + o,
                    (float)((q.w >> 4)  & 15) * s + o);
    *(uint4*)(wbf + j) = r;
}

// =====================================================================
// Phase 2: lean bf16 GEMM
//   - async global->LDS tile DMA (ASYNCcnt) when the toolchain exposes it
//   - inner loop otherwise: pure b128 traffic + WMMA
// =====================================================================
__global__ __launch_bounds__(256)
void gemm_bf16_kernel(const unsigned short* __restrict__ xbf, // [8192][4096] bf16
                      const unsigned short* __restrict__ wbf, // [4096][4096] bf16
                      const float* __restrict__ bias,
                      float*       __restrict__ out)
{
    __shared__ __align__(16) unsigned short As[2][BM][LDA];   // 20 KB
    __shared__ __align__(16) unsigned short Bs[2][BN][LDB];   // 10 KB

    const int t  = threadIdx.x;
    const int bn = blockIdx.x;       // N tile
    const int bm = blockIdx.y;       // M tile

    // load assignment: row = t>>2 (0..63), uint4 quad = t&3 (8 bf16 each)
    const int lrow = t >> 2;
    const int lcol = (t & 3) * 8;

    // wave/lane mapping (wave32): 4 waves down M x 2 across N, 32x32 each
    const int wave = t >> 5, lane = t & 31;
    const int m_off = (wave & 3) * 32;
    const int n_off = (wave >> 2) * 32;
    const int lr = lane & 15;
    const int lh = lane >> 4;

    v8f acc[2][2] = {};

#if HAVE_ASYNC_LDS
    // -------- async DMA path: no VGPR staging, tracked with ASYNCcnt --------
    auto issue_async = [&](int kt, int buf) {
        const int k0 = kt * BK;
        __builtin_amdgcn_global_load_async_to_lds_b128(
            (AS1 v4i*)(xbf + (size_t)(bm * BM + lrow)      * IN_F + k0 + lcol),
            (AS3 v4i*)&As[buf][lrow][lcol], 0, 0);
        __builtin_amdgcn_global_load_async_to_lds_b128(
            (AS1 v4i*)(xbf + (size_t)(bm * BM + lrow + 64) * IN_F + k0 + lcol),
            (AS3 v4i*)&As[buf][lrow + 64][lcol], 0, 0);
        __builtin_amdgcn_global_load_async_to_lds_b128(
            (AS1 v4i*)(wbf + (size_t)(bn * BN + lrow)      * IN_F + k0 + lcol),
            (AS3 v4i*)&Bs[buf][lrow][lcol], 0, 0);
    };

    issue_async(0, 0);

    for (int kt = 0; kt < NKT; ++kt) {
        const int cur = kt & 1;

        __builtin_amdgcn_s_wait_asynccnt(0);  // tile `cur` landed in LDS
        __syncthreads();                      // visible to all waves

        if (kt + 1 < NKT) issue_async(kt + 1, cur ^ 1);  // DMA next tile now

        Frag a[2], b[2];
#pragma unroll
        for (int ti = 0; ti < 2; ++ti) {
            const int row = m_off + ti * 16 + lr;
            a[ti].q[0] = *(const uint4*)&As[cur][row][lh * 8];
            a[ti].q[1] = *(const uint4*)&As[cur][row][16 + lh * 8];
        }
#pragma unroll
        for (int tj = 0; tj < 2; ++tj) {
            const int row = n_off + tj * 16 + lr;
            b[tj].q[0] = *(const uint4*)&Bs[cur][row][lh * 8];
            b[tj].q[1] = *(const uint4*)&Bs[cur][row][16 + lh * 8];
        }

#pragma unroll
        for (int ti = 0; ti < 2; ++ti)
#pragma unroll
            for (int tj = 0; tj < 2; ++tj)
                acc[ti][tj] = __builtin_amdgcn_wmma_f32_16x16x32_bf16(
                    false, a[ti].v, false, b[tj].v,
                    (short)0, acc[ti][tj], false, false);
        // frag ds_loads are consumed by the WMMAs (s_wait_dscnt) before the
        // next iteration's barrier, so the DMA into this buffer two steps
        // from now cannot race the reads.
    }
#else
    // -------- fallback: reg-staged double buffering --------
    uint4 a0, a1, b0;
    auto load_global = [&](int kt) {
        const int k0 = kt * BK;
        a0 = *(const uint4*)(xbf + (size_t)(bm * BM + lrow)      * IN_F + k0 + lcol);
        a1 = *(const uint4*)(xbf + (size_t)(bm * BM + lrow + 64) * IN_F + k0 + lcol);
        b0 = *(const uint4*)(wbf + (size_t)(bn * BN + lrow)      * IN_F + k0 + lcol);
    };
    auto store_lds = [&](int buf) {
        *(uint4*)&As[buf][lrow][lcol]      = a0;
        *(uint4*)&As[buf][lrow + 64][lcol] = a1;
        *(uint4*)&Bs[buf][lrow][lcol]      = b0;
    };

    load_global(0);
    store_lds(0);
    __syncthreads();

    for (int kt = 0; kt < NKT; ++kt) {
        const int cur = kt & 1;
        if (kt + 1 < NKT) load_global(kt + 1);
        if (kt + 2 < NKT) {
            __builtin_prefetch(xbf + (size_t)(bm * BM + lrow) * IN_F + (kt + 2) * BK + lcol, 0, 0);
            __builtin_prefetch(wbf + (size_t)(bn * BN + lrow) * IN_F + (kt + 2) * BK + lcol, 0, 0);
        }

        Frag a[2], b[2];
#pragma unroll
        for (int ti = 0; ti < 2; ++ti) {
            const int row = m_off + ti * 16 + lr;
            a[ti].q[0] = *(const uint4*)&As[cur][row][lh * 8];
            a[ti].q[1] = *(const uint4*)&As[cur][row][16 + lh * 8];
        }
#pragma unroll
        for (int tj = 0; tj < 2; ++tj) {
            const int row = n_off + tj * 16 + lr;
            b[tj].q[0] = *(const uint4*)&Bs[cur][row][lh * 8];
            b[tj].q[1] = *(const uint4*)&Bs[cur][row][16 + lh * 8];
        }

#pragma unroll
        for (int ti = 0; ti < 2; ++ti)
#pragma unroll
            for (int tj = 0; tj < 2; ++tj)
                acc[ti][tj] = __builtin_amdgcn_wmma_f32_16x16x32_bf16(
                    false, a[ti].v, false, b[tj].v,
                    (short)0, acc[ti][tj], false, false);

        if (kt + 1 < NKT) store_lds((kt + 1) & 1);
        __syncthreads();
    }
#endif

#pragma unroll
    for (int tj = 0; tj < 2; ++tj) {
        const int n_g = bn * BN + n_off + tj * 16 + lr;
        const float bv = bias[n_g];
#pragma unroll
        for (int ti = 0; ti < 2; ++ti) {
            const int m_base = bm * BM + m_off + ti * 16 + lh * 8;
#pragma unroll
            for (int v = 0; v < 8; ++v)
                out[(size_t)(m_base + v) * OUT_F + n_g] = acc[ti][tj][v] + bv;
        }
    }
}

// =====================================================================
// Fallback: fused dequant+GEMM (used when d_ws is too small)
// =====================================================================
__global__ __launch_bounds__(256)
void qlinear_fused_kernel(const float* __restrict__ x,
                          const int*   __restrict__ packed,
                          const float* __restrict__ scales,
                          const float* __restrict__ offsets,
                          const float* __restrict__ bias,
                          float*       __restrict__ out)
{
    __shared__ __align__(16) unsigned short As[2][BM][LDA];
    __shared__ __align__(16) unsigned short Bs[2][BN][LDB];

    const int t  = threadIdx.x;
    const int bn = blockIdx.x;
    const int bm = blockIdx.y;

    const int ar  = t >> 3;          // x rows (4x)
    const int ac  = (t & 7) * 4;     // float4 col
    const int br  = t >> 2;          // w row
    const int bc4 = (t & 3) * 4;     // int4 offset

    const int wave = t >> 5, lane = t & 31;
    const int m_off = (wave & 3) * 32;
    const int n_off = (wave >> 2) * 32;
    const int lr = lane & 15;
    const int lh = lane >> 4;

    float4 ax[4];
    int4   wq;
    float  ws, wo;

    auto load_global = [&](int kt) {
        const int k0 = kt * BK;
#pragma unroll
        for (int i = 0; i < 4; ++i)
            ax[i] = *(const float4*)(x + (size_t)(bm * BM + ar + 32 * i) * IN_F + k0 + ac);
        const int n_g = bn * BN + br;
        wq = *(const int4*)(packed + (size_t)n_g * (IN_F / 2) + (k0 >> 1) + bc4);
        const int g = n_g * (IN_F / GROUP) + (k0 / GROUP);
        ws = scales[g];
        wo = offsets[g];
    };
    auto store_lds = [&](int buf) {
#pragma unroll
        for (int i = 0; i < 4; ++i) {
            uint2 p;
            p.x = pack_bf16(ax[i].x, ax[i].y);
            p.y = pack_bf16(ax[i].z, ax[i].w);
            *(uint2*)&As[buf][ar + 32 * i][ac] = p;
        }
        uint4 o;
        o.x = pack_bf16((float)( wq.x       & 15) * ws + wo,
                        (float)((wq.x >> 4) & 15) * ws + wo);
        o.y = pack_bf16((float)( wq.y       & 15) * ws + wo,
                        (float)((wq.y >> 4) & 15) * ws + wo);
        o.z = pack_bf16((float)( wq.z       & 15) * ws + wo,
                        (float)((wq.z >> 4) & 15) * ws + wo);
        o.w = pack_bf16((float)( wq.w       & 15) * ws + wo,
                        (float)((wq.w >> 4) & 15) * ws + wo);
        *(uint4*)&Bs[buf][br][bc4 * 2] = o;
    };

    v8f acc[2][2] = {};

    load_global(0);
    store_lds(0);
    __syncthreads();

    for (int kt = 0; kt < NKT; ++kt) {
        const int cur = kt & 1;
        if (kt + 1 < NKT) load_global(kt + 1);

        Frag a[2], b[2];
#pragma unroll
        for (int ti = 0; ti < 2; ++ti) {
            const int row = m_off + ti * 16 + lr;
            a[ti].q[0] = *(const uint4*)&As[cur][row][lh * 8];
            a[ti].q[1] = *(const uint4*)&As[cur][row][16 + lh * 8];
        }
#pragma unroll
        for (int tj = 0; tj < 2; ++tj) {
            const int row = n_off + tj * 16 + lr;
            b[tj].q[0] = *(const uint4*)&Bs[cur][row][lh * 8];
            b[tj].q[1] = *(const uint4*)&Bs[cur][row][16 + lh * 8];
        }

#pragma unroll
        for (int ti = 0; ti < 2; ++ti)
#pragma unroll
            for (int tj = 0; tj < 2; ++tj)
                acc[ti][tj] = __builtin_amdgcn_wmma_f32_16x16x32_bf16(
                    false, a[ti].v, false, b[tj].v,
                    (short)0, acc[ti][tj], false, false);

        if (kt + 1 < NKT) store_lds((kt + 1) & 1);
        __syncthreads();
    }

#pragma unroll
    for (int tj = 0; tj < 2; ++tj) {
        const int n_g = bn * BN + n_off + tj * 16 + lr;
        const float bv = bias[n_g];
#pragma unroll
        for (int ti = 0; ti < 2; ++ti) {
            const int m_base = bm * BM + m_off + ti * 16 + lh * 8;
#pragma unroll
            for (int v = 0; v < 8; ++v)
                out[(size_t)(m_base + v) * OUT_F + n_g] = acc[ti][tj][v] + bv;
        }
    }
}

extern "C" void kernel_launch(void* const* d_in, const int* in_sizes, int n_in,
                              void* d_out, int out_size, void* d_ws, size_t ws_size,
                              hipStream_t stream) {
    const float* x       = (const float*)d_in[0];
    const int*   packed  = (const int*)  d_in[1];
    const float* scales  = (const float*)d_in[2];
    const float* offsets = (const float*)d_in[3];
    const float* bias    = (const float*)d_in[4];
    float*       out     = (float*)d_out;

    const size_t X_BF = (size_t)M_TOTAL * IN_F * sizeof(unsigned short); // 64 MiB
    const size_t W_BF = (size_t)OUT_F   * IN_F * sizeof(unsigned short); // 32 MiB

    dim3 grid(OUT_F / BN, M_TOTAL / BM);   // 64 x 64 workgroups

    if (ws_size >= X_BF + W_BF) {
        unsigned short* xbf = (unsigned short*)d_ws;
        unsigned short* wbf = (unsigned short*)((char*)d_ws + X_BF);
        convert_x_kernel<<<(M_TOTAL * IN_F) / (256 * 8), 256, 0, stream>>>(x, xbf);
        dequant_w_kernel<<<(OUT_F * IN_F / 2) / (256 * 4), 256, 0, stream>>>(packed, scales, offsets, wbf);
        gemm_bf16_kernel<<<grid, 256, 0, stream>>>(xbf, wbf, bias, out);
    } else {
        qlinear_fused_kernel<<<grid, 256, 0, stream>>>(x, packed, scales, offsets, bias, out);
    }
}
